// Transformer_15599321219504
// MI455X (gfx1250) — compile-verified, matrix operand order checked by reading, not run on hardware
//
#include <hip/hip_runtime.h>
#include <hip/hip_bf16.h>

// ---------------- problem constants ----------------
#define LL 4
#define DD 256
#define HH 8
#define MM 1024
#define BB 8
#define NN 1024
#define HDIM 2048          // H * D
#define BN (BB * NN)       // 8192 rows

typedef __bf16 bf16_t;
typedef __attribute__((ext_vector_type(16))) __bf16 v16bf;
typedef __attribute__((ext_vector_type(8)))  float  v8f;
typedef __attribute__((ext_vector_type(4)))  int    v4i_t;

// ---------------- async global->LDS staging (CDNA5) ----------------
#if defined(__has_builtin)
#if __has_builtin(__builtin_amdgcn_global_load_async_to_lds_b128) && \
    __has_builtin(__builtin_amdgcn_s_wait_asynccnt)
#define USE_ASYNC_LDS 1
#endif
#endif

__device__ inline void stage16(const bf16_t* gsrc, bf16_t* ldst) {
#ifdef USE_ASYNC_LDS
  // global_load_async_to_lds_b128 (ASYNCcnt-tracked DMA into LDS)
  __builtin_amdgcn_global_load_async_to_lds_b128(
      (__attribute__((address_space(1))) v4i_t*)gsrc,
      (__attribute__((address_space(3))) v4i_t*)ldst, 0, 0);
#else
  *(uint4*)ldst = *(const uint4*)gsrc;
#endif
}
__device__ inline void stage_fence() {
#ifdef USE_ASYNC_LDS
  __builtin_amdgcn_s_wait_asynccnt(0);
#endif
}

// ---------------- WMMA helpers (CDNA5, wave32) ----------------
__device__ inline v8f wmma_bf16(v16bf a, v16bf b, v8f c) {
  return __builtin_amdgcn_wmma_f32_16x16x32_bf16(
      /*neg_a=*/false, a, /*neg_b=*/false, b,
      /*c_mod=*/(short)0, c, /*reuse_a=*/false, /*reuse_b=*/false);
}

// A fragment: 16x32 (MxK), row-major src, element (r, k). ISA 7.12.2 layout.
// Per lane: two contiguous 8-element (16B) runs -> 2 x b128 loads.
__device__ inline v16bf load_frag_a(const bf16_t* __restrict__ src, int ld) {
  int lane = threadIdx.x & 31;
  int half = lane >> 4, r = lane & 15;
  v16bf a;
#pragma unroll
  for (int i = 0; i < 8; ++i) {
    int kk = (i >> 2) * 16 + half * 8 + (i & 3) * 2;
    a[2 * i]     = src[r * ld + kk];
    a[2 * i + 1] = src[r * ld + kk + 1];
  }
  return a;
}

// B fragment: 32x16 (KxN) from K-CONTIGUOUS storage: element (k,c)=src[c*ld+k].
// Per lane: two contiguous 16B runs -> 2 x b128 loads.
__device__ inline v16bf load_frag_bt(const bf16_t* __restrict__ src, int ld) {
  int lane = threadIdx.x & 31;
  int half = lane >> 4, c = lane & 15;
  v16bf b;
#pragma unroll
  for (int i = 0; i < 8; ++i) {
    int kk = (i >> 2) * 16 + half * 8 + (i & 3) * 2;
    b[2 * i]     = src[c * ld + kk];
    b[2 * i + 1] = src[c * ld + kk + 1];
  }
  return b;
}

// ---------------- fp32 -> bf16 convert (+ transpose for weights) ----------
__global__ void cvt_bf16_kernel(const float* __restrict__ in,
                                bf16_t* __restrict__ out, int n) {
  int i = blockIdx.x * blockDim.x + threadIdx.x;
  if (i < n) out[i] = (bf16_t)in[i];
}

// in: [L][K][NC] fp32 -> out: [L][NC][K] bf16  (K-contiguous weights)
__global__ void cvt_t_kernel(const float* __restrict__ in,
                             bf16_t* __restrict__ out, int K, int NC,
                             int total) {
  int i = blockIdx.x * blockDim.x + threadIdx.x;
  if (i >= total) return;
  int kn = K * NC;
  int l = i / kn, r = i % kn;
  int k = r / NC, c = r % NC;
  out[(size_t)l * kn + (size_t)c * K + k] = (bf16_t)in[i];
}

// ---------------- LayerNorm (row = D=256, one block/row) ----------------
__global__ __launch_bounds__(256) void ln_kernel(const float* __restrict__ x,
                                                 const float* __restrict__ g,
                                                 const float* __restrict__ bia,
                                                 bf16_t* __restrict__ y) {
  __shared__ float red[256];
  int row = blockIdx.x, t = threadIdx.x;
  float v = x[(size_t)row * DD + t];
  red[t] = v;
  __syncthreads();
#pragma unroll
  for (int s = 128; s > 0; s >>= 1) {
    if (t < s) red[t] += red[t + s];
    __syncthreads();
  }
  float mean = red[0] * (1.0f / DD);
  __syncthreads();
  float dv = v - mean;
  red[t] = dv * dv;
  __syncthreads();
#pragma unroll
  for (int s = 128; s > 0; s >>= 1) {
    if (t < s) red[t] += red[t + s];
    __syncthreads();
  }
  float var = red[0] * (1.0f / DD);
  float r = rsqrtf(var + 1e-5f);
  y[(size_t)row * DD + t] = (bf16_t)(dv * r * g[t] + bia[t]);
}

// ---------------- tiled WMMA GEMM, 128x128x32, 8 waves ----------------
// W is pre-transposed: Wt[NC][K] (K-contiguous).
// MODE 0: qkv scatter (bias; q,k -> [B,H,N,D]; v -> [B,H,D,N])
// MODE 1: residual add fp32 (out = resid_in + acc + bias)
// MODE 2: exact GELU, bf16 out
template <int MODE>
__global__ __launch_bounds__(256) void gemm_kernel(
    const bf16_t* __restrict__ A, const bf16_t* __restrict__ Wt,
    const float* __restrict__ bias, int K, int NC,
    const float* __restrict__ resid_in, float* __restrict__ resid_out,
    bf16_t* __restrict__ out_bf, bf16_t* __restrict__ qb,
    bf16_t* __restrict__ kb, bf16_t* __restrict__ vb) {
  __shared__ __align__(16) bf16_t As[128 * 32];   // [row][k]
  __shared__ __align__(16) bf16_t Bt[128 * 32];   // [col][k]

  int tid = threadIdx.x;
  int wid = tid >> 5, lane = tid & 31;
  int wm = wid >> 2, wn = wid & 3;           // 2 x 4 wave grid
  int row0 = blockIdx.y * 128;
  int col0 = blockIdx.x * 128;

  v8f acc[4][2] = {};

  for (int k0 = 0; k0 < K; k0 += 32) {
    // stage A: 128 rows x 32 k, and Bt: 128 cols x 32 k (both coalesced 16B)
    for (int idx = tid; idx < (128 * 32) / 8; idx += 256) {
      int r = idx >> 2, c8 = (idx & 3) * 8;
      stage16(&A[(size_t)(row0 + r) * K + k0 + c8], &As[r * 32 + c8]);
      stage16(&Wt[(size_t)(col0 + r) * K + k0 + c8], &Bt[r * 32 + c8]);
    }
    if (k0 + 32 < K)
      __builtin_prefetch(&Wt[(size_t)col0 * K + k0 + 32], 0, 1);
    stage_fence();
    __syncthreads();

    v16bf bf0 = load_frag_bt(&Bt[(wn * 32 + 0) * 32], 32);
    v16bf bf1 = load_frag_bt(&Bt[(wn * 32 + 16) * 32], 32);
#pragma unroll
    for (int mi = 0; mi < 4; ++mi) {
      v16bf af = load_frag_a(&As[(wm * 64 + mi * 16) * 32], 32);
      acc[mi][0] = wmma_bf16(af, bf0, acc[mi][0]);
      acc[mi][1] = wmma_bf16(af, bf1, acc[mi][1]);
    }
    __syncthreads();
  }

  // epilogue (C layout: lane<16 rows v, lane>=16 rows v+8; col = lane&15)
  int half = lane >> 4, lc = lane & 15;
#pragma unroll
  for (int mi = 0; mi < 4; ++mi) {
#pragma unroll
    for (int ni = 0; ni < 2; ++ni) {
#pragma unroll
      for (int vv = 0; vv < 8; ++vv) {
        int gr = row0 + wm * 64 + mi * 16 + half * 8 + vv;
        int gc = col0 + wn * 32 + ni * 16 + lc;
        float val = acc[mi][ni][vv] + bias[gc];
        if (MODE == 0) {
          // gc = h*768 + d*3 + s  (reshape 'b n (h d 3)')
          int s = gc % 3;
          int dd2 = (gc / 3) & (DD - 1);
          int hh2 = gc / (3 * DD);
          int b2 = gr >> 10, n2 = gr & (NN - 1);
          bf16_t ov = (bf16_t)val;
          if (s == 0) {
            qb[(((size_t)(b2 * HH + hh2) * NN + n2) * DD) + dd2] = ov;
          } else if (s == 1) {
            kb[(((size_t)(b2 * HH + hh2) * NN + n2) * DD) + dd2] = ov;
          } else {  // V stored d-major: [B,H,D,N]
            vb[(((size_t)(b2 * HH + hh2) * DD + dd2) * NN) + n2] = ov;
          }
        } else if (MODE == 1) {
          size_t off = (size_t)gr * NC + gc;
          resid_out[off] = resid_in[off] + val;
        } else {
          float ge = 0.5f * val * (1.0f + erff(val * 0.70710678118654752f));
          out_bf[(size_t)gr * NC + gc] = (bf16_t)ge;
        }
      }
    }
  }
}

// ---------------- flash attention ----------------
// grid: (N/128, B*H); 256 threads = 8 waves; each wave owns 16 q-rows.
// q,k: [B,H,N,D]; v: [B,H,D,N] (d-major so V fragments are key-contiguous).
__global__ __launch_bounds__(256) void attn_kernel(
    const bf16_t* __restrict__ q, const bf16_t* __restrict__ k,
    const bf16_t* __restrict__ v, bf16_t* __restrict__ o) {
  __shared__ __align__(16) bf16_t Ks[32 * 256];   // [key][d]
  __shared__ __align__(16) bf16_t Vst[256 * 32];  // [d][key]
  __shared__ __align__(16) bf16_t Ps[8][16 * 32];

  int bh = blockIdx.y;
  int tid = threadIdx.x, wid = tid >> 5, lane = tid & 31;
  int half = lane >> 4, lc = lane & 15;
  const bf16_t* Qb = q + (size_t)bh * NN * DD;
  const bf16_t* Kb = k + (size_t)bh * NN * DD;
  const bf16_t* Vb = v + (size_t)bh * DD * NN;
  int qrow0 = blockIdx.x * 128 + wid * 16;

  // resident Q fragments: 16 rows x 256 (8 x K=32 fragments)
  v16bf qf[8];
#pragma unroll
  for (int kk = 0; kk < 8; ++kk)
    qf[kk] = load_frag_a(Qb + (size_t)qrow0 * DD + kk * 32, DD);

  v8f Oacc[16] = {};
  float mrow[8], lrow[8];
#pragma unroll
  for (int vv = 0; vv < 8; ++vv) { mrow[vv] = -1e30f; lrow[vv] = 0.0f; }

  for (int c0 = 0; c0 < NN; c0 += 32) {
    // stage K chunk [32 keys][256 d] and V chunk [256 d][32 keys]
    for (int idx = tid; idx < (32 * 256) / 8; idx += 256) {
      int r = idx >> 5, c8 = (idx & 31) * 8;
      stage16(&Kb[(size_t)(c0 + r) * DD + c8], &Ks[r * 256 + c8]);
      int d = idx >> 2, k8 = (idx & 3) * 8;
      stage16(&Vb[(size_t)d * NN + c0 + k8], &Vst[d * 32 + k8]);
    }
    stage_fence();
    __syncthreads();

    // S = Q * K^T  (16 q-rows x 32 keys, two 16x16 tiles)
    v8f S[2] = {};
#pragma unroll
    for (int t = 0; t < 2; ++t)
#pragma unroll
      for (int kk = 0; kk < 8; ++kk)
        S[t] = wmma_bf16(qf[kk],
                         load_frag_bt(&Ks[(t * 16) * 256 + kk * 32], 256),
                         S[t]);

    // online softmax: row reductions across the 16-lane column group
    float mnew[8], alpha[8];
#pragma unroll
    for (int vv = 0; vv < 8; ++vv) {
      float mx = fmaxf(S[0][vv], S[1][vv]);
#pragma unroll
      for (int off = 1; off < 16; off <<= 1)
        mx = fmaxf(mx, __shfl_xor(mx, off, 32));
      mnew[vv] = fmaxf(mrow[vv], mx);
      alpha[vv] = __expf(mrow[vv] - mnew[vv]);
      mrow[vv] = mnew[vv];
    }
#pragma unroll
    for (int vv = 0; vv < 8; ++vv) {
      float p0 = __expf(S[0][vv] - mnew[vv]);
      float p1 = __expf(S[1][vv] - mnew[vv]);
      S[0][vv] = p0;
      S[1][vv] = p1;
      float sum = p0 + p1;
#pragma unroll
      for (int off = 1; off < 16; off <<= 1)
        sum += __shfl_xor(sum, off, 32);
      lrow[vv] = lrow[vv] * alpha[vv] + sum;
    }
#pragma unroll
    for (int dt = 0; dt < 16; ++dt)
#pragma unroll
      for (int vv = 0; vv < 8; ++vv) Oacc[dt][vv] *= alpha[vv];

    // P (C layout) -> bf16 -> LDS -> A-fragment layout
    bf16_t* Pl = &Ps[wid][0];
#pragma unroll
    for (int t = 0; t < 2; ++t)
#pragma unroll
      for (int vv = 0; vv < 8; ++vv)
        Pl[(half * 8 + vv) * 32 + t * 16 + lc] = (bf16_t)S[t][vv];
    __syncthreads();

    v16bf pf = load_frag_a(Pl, 32);
#pragma unroll
    for (int dt = 0; dt < 16; ++dt)
      Oacc[dt] =
          wmma_bf16(pf, load_frag_bt(&Vst[(dt * 16) * 32], 32), Oacc[dt]);
    __syncthreads();
  }

  // epilogue: O / (l * sqrt(D)); scale=16 applied post-softmax (as reference)
  int b2 = bh >> 3, h2 = bh & 7;
#pragma unroll
  for (int vv = 0; vv < 8; ++vv) {
    float inv = 1.0f / (lrow[vv] * 16.0f);
    int n2 = qrow0 + half * 8 + vv;
    size_t base = ((size_t)(b2 * NN + n2) * HDIM) + (size_t)h2 * DD;
#pragma unroll
    for (int dt = 0; dt < 16; ++dt)
      o[base + dt * 16 + lc] = (bf16_t)(Oacc[dt][vv] * inv);
  }
}

// ---------------- host side ----------------
extern "C" void kernel_launch(void* const* d_in, const int* in_sizes, int n_in,
                              void* d_out, int out_size, void* d_ws,
                              size_t ws_size, hipStream_t stream) {
  const float* x_in   = (const float*)d_in[0];
  const float* ln1_g  = (const float*)d_in[1];
  const float* ln1_b  = (const float*)d_in[2];
  const float* qkv_w  = (const float*)d_in[3];
  const float* qkv_b  = (const float*)d_in[4];
  const float* proj_w = (const float*)d_in[5];
  const float* proj_b = (const float*)d_in[6];
  const float* ln2_g  = (const float*)d_in[7];
  const float* ln2_b  = (const float*)d_in[8];
  const float* w1     = (const float*)d_in[9];
  const float* b1     = (const float*)d_in[10];
  const float* w2     = (const float*)d_in[11];
  const float* b2     = (const float*)d_in[12];

  char* wsp = (char*)d_ws;
  size_t off = 0;
  auto alloc = [&](size_t bytes) -> void* {
    void* p = wsp + off;
    off = (off + bytes + 255) & ~(size_t)255;
    return p;
  };

  float*  x     = (float*)alloc((size_t)BN * DD * 4);
  bf16_t* h     = (bf16_t*)alloc((size_t)BN * DD * 2);
  bf16_t* qkvwt = (bf16_t*)alloc((size_t)LL * DD * 3 * HDIM * 2);
  bf16_t* projwt= (bf16_t*)alloc((size_t)LL * HDIM * DD * 2);
  bf16_t* w1t   = (bf16_t*)alloc((size_t)LL * DD * MM * 2);
  bf16_t* w2t   = (bf16_t*)alloc((size_t)LL * MM * DD * 2);
  bf16_t* qb    = (bf16_t*)alloc((size_t)BB * HH * NN * DD * 2);
  bf16_t* kb    = (bf16_t*)alloc((size_t)BB * HH * NN * DD * 2);
  bf16_t* vb    = (bf16_t*)alloc((size_t)BB * HH * NN * DD * 2);
  bf16_t* ob    = (bf16_t*)alloc((size_t)BN * HDIM * 2);
  bf16_t* hid   = (bf16_t*)alloc((size_t)BN * MM * 2);

  // weights -> bf16, transposed to [L][NC][K]
  int n_qkvw = LL * DD * 3 * HDIM;
  int n_projw = LL * HDIM * DD;
  int n_w1 = LL * DD * MM, n_w2 = LL * MM * DD;
  cvt_t_kernel<<<(n_qkvw + 255) / 256, 256, 0, stream>>>(qkv_w, qkvwt, DD,
                                                         3 * HDIM, n_qkvw);
  cvt_t_kernel<<<(n_projw + 255) / 256, 256, 0, stream>>>(proj_w, projwt, HDIM,
                                                          DD, n_projw);
  cvt_t_kernel<<<(n_w1 + 255) / 256, 256, 0, stream>>>(w1, w1t, DD, MM, n_w1);
  cvt_t_kernel<<<(n_w2 + 255) / 256, 256, 0, stream>>>(w2, w2t, MM, DD, n_w2);

  (void)hipMemcpyAsync(x, x_in, (size_t)BN * DD * 4, hipMemcpyDeviceToDevice,
                       stream);

  for (int i = 0; i < LL; ++i) {
    // --- attention block ---
    ln_kernel<<<BN, 256, 0, stream>>>(x, ln1_g + i * DD, ln1_b + i * DD, h);
    gemm_kernel<0><<<dim3((3 * HDIM) / 128, BN / 128), 256, 0, stream>>>(
        h, qkvwt + (size_t)i * DD * 3 * HDIM, qkv_b + (size_t)i * 3 * HDIM,
        DD, 3 * HDIM, nullptr, nullptr, nullptr, qb, kb, vb);
    attn_kernel<<<dim3(NN / 128, BB * HH), 256, 0, stream>>>(qb, kb, vb, ob);
    gemm_kernel<1><<<dim3(DD / 128, BN / 128), 256, 0, stream>>>(
        ob, projwt + (size_t)i * HDIM * DD, proj_b + (size_t)i * DD, HDIM, DD,
        x, x, nullptr, nullptr, nullptr, nullptr);
    // --- MLP block ---
    ln_kernel<<<BN, 256, 0, stream>>>(x, ln2_g + i * DD, ln2_b + i * DD, h);
    gemm_kernel<2><<<dim3(MM / 128, BN / 128), 256, 0, stream>>>(
        h, w1t + (size_t)i * DD * MM, b1 + (size_t)i * MM, DD, MM,
        nullptr, nullptr, hid, nullptr, nullptr, nullptr);
    gemm_kernel<1><<<dim3(DD / 128, BN / 128), 256, 0, stream>>>(
        hid, w2t + (size_t)i * MM * DD, b2 + (size_t)i * DD, MM, DD,
        x, x, nullptr, nullptr, nullptr, nullptr);
  }

  (void)hipMemcpyAsync(d_out, x, (size_t)BN * DD * 4,
                       hipMemcpyDeviceToDevice, stream);
}
